// CFConv_24043226923283
// MI455X (gfx1250) — matrix operand order repaired and checked by reading, not compiled, not run on hardware
//
#include <hip/hip_runtime.h>
#include <hip/hip_bf16.h>

// CFConv (SchNet) for MI455X / gfx1250, wave32 + v_wmma_f32_16x16x32_bf16.
// B=32, A=256, N=64, G=50, F=128.

#define Bdim 32
#define Adim 256
#define Ndim 64
#define Gdim 50
#define Fdim 128

typedef __attribute__((ext_vector_type(16))) __bf16 v16bf;
typedef __attribute__((ext_vector_type(8)))  float  v8f;

// ---------------- WMMA fragment helpers (wave32 layouts per CDNA5 ISA) ----

// A-matrix 16x32 bf16, row-major source with leading dim `lda` (halfs).
// Lane l: row m = l&15, half h = l>>4. Elements 0..7 -> K = kOff + h*8 + {0..7},
// elements 8..15 -> K = kOff + h*8 + 16 + {0..7}.
__device__ __forceinline__ v16bf load_frag_a(const __bf16* base, int row0,
                                             int lda, int kOff, int lane) {
  const int m = lane & 15, h = lane >> 4;
  const __bf16* p = base + (row0 + m) * lda + kOff + h * 8;
  v16bf f;
#pragma unroll
  for (int i = 0; i < 8; ++i) f[i] = p[i];
#pragma unroll
  for (int i = 0; i < 8; ++i) f[8 + i] = p[16 + i];
  return f;
}

// B-matrix 32x16 bf16 taken from an N-major (transposed) buffer: bt[n][k],
// leading dim `ldb` (halfs). Lane l: col n = l&15, half h = l>>4,
// elements 0..7 -> K = kOff + h*8 + {0..7}, 8..15 -> K = kOff + h*8 + 16 + {0..7}.
__device__ __forceinline__ v16bf load_frag_bt(const __bf16* bt, int col0,
                                              int ldb, int kOff, int lane) {
  const int n = lane & 15, h = lane >> 4;
  const __bf16* p = bt + (col0 + n) * ldb + kOff + h * 8;
  v16bf f;
#pragma unroll
  for (int i = 0; i < 8; ++i) f[i] = p[i];
#pragma unroll
  for (int i = 0; i < 8; ++i) f[8 + i] = p[16 + i];
  return f;
}

__device__ __forceinline__ v8f wmma_bf16(v16bf a, v16bf b, v8f c) {
  return __builtin_amdgcn_wmma_f32_16x16x32_bf16(
      /*neg_a=*/false, a, /*neg_b=*/false, b,
      /*c_mod=*/(short)0, c, /*reuse_a=*/false, /*reuse_b=*/false);
}

// Numerically stable shifted softplus: log(1+e^v) - log(2)
__device__ __forceinline__ float ssp(float v) {
  return fmaxf(v, 0.0f) + log1pf(expf(-fabsf(v))) - 0.69314718056f;
}

// ---------------- Kernel 1: y = x @ in2f_w  (8192x128 @ 128x128) ----------

__global__ __launch_bounds__(256) void in2f_gemm_kernel(
    const float* __restrict__ x, const float* __restrict__ w,
    float* __restrict__ y) {
  __shared__ __bf16 sA[64 * Fdim];        // 64x128 block of x (bf16), 16 KB
  __shared__ __bf16 sBT[Fdim * Fdim];     // w transposed [f][k], 32 KB

  const int tid  = threadIdx.x;
  const int row0 = blockIdx.x * 64;       // 128 blocks cover 8192 rows

  for (int i = tid; i < 64 * Fdim; i += 256) {
    const int r = i >> 7, k = i & 127;
    sA[i] = (__bf16)x[(size_t)(row0 + r) * Fdim + k];
  }
  for (int i = tid; i < Fdim * Fdim; i += 256) {
    const int k = i >> 7, f = i & 127;    // w is (k, f) row-major
    sBT[f * Fdim + k] = (__bf16)w[i];
  }
  __syncthreads();

  const int wv = tid >> 5, lane = tid & 31;
  const int tf = wv;                      // wave owns column tile tf
  const int n = lane & 15, h = lane >> 4;

#pragma unroll
  for (int tm = 0; tm < 4; ++tm) {
    v8f acc = {};
#pragma unroll
    for (int kp = 0; kp < 4; ++kp) {
      v16bf a = load_frag_a(sA, 16 * tm, Fdim, 32 * kp, lane);
      v16bf b = load_frag_bt(sBT, 16 * tf, Fdim, 32 * kp, lane);
      acc = wmma_bf16(a, b, acc);
    }
#pragma unroll
    for (int r = 0; r < 8; ++r) {
      const int m = 16 * tm + r + 8 * h;
      y[(size_t)(row0 + m) * Fdim + 16 * tf + n] = acc[r];
    }
  }
}

// ---------------- Kernel 2: fused filter-net + cutoff + gather + reduce ---
// One workgroup (256 thr / 8 waves) per (b,a) pair.

__global__ __launch_bounds__(256) void cfconv_kernel(
    const float* __restrict__ fij, const float* __restrict__ rij,
    const float* __restrict__ mask, const long long* __restrict__ nbr,
    const float* __restrict__ fw1, const float* __restrict__ fb1,
    const float* __restrict__ fw2, const float* __restrict__ fb2,
    const float* __restrict__ y, float* __restrict__ out) {
  __shared__ __bf16 sA[Ndim * 64];        // f_ij zero-padded 64x64, 8 KB
  __shared__ __bf16 sBT[Fdim * Fdim];     // fw1^T (64-wide) then fw2^T, 32 KB
  __shared__ __bf16 sH[Ndim * Fdim];      // ssp(GEMM1) 64x128 bf16, 16 KB
  __shared__ float  sC[Ndim];             // cutoff * mask
  __shared__ int    sN[Ndim];             // neighbor indices
  __shared__ float  sOut[Fdim];           // fp32 reduction buffer

  const int tid = threadIdx.x;
  const int p   = blockIdx.x;             // pair index = b*A + a
  const int bb  = p >> 8;                 // batch (A == 256)

  if (tid < Fdim) sOut[tid] = 0.0f;
  if (tid < Ndim) {
    const float r = rij[(size_t)p * Ndim + tid];
    float c = 0.5f * (cosf(r * 0.628318530718f) + 1.0f);   // pi/5
    c = (r < 5.0f) ? c : 0.0f;
    sC[tid] = c * mask[(size_t)p * Ndim + tid];
    sN[tid] = (int)nbr[(size_t)p * Ndim + tid];            // int64 input
  }
  // f_ij (64x50) -> sA zero-padded to 64x64
  for (int i = tid; i < Ndim * 64; i += 256) {
    const int row = i >> 6, g = i & 63;
    sA[i] = (__bf16)((g < Gdim) ? fij[(size_t)p * Ndim * Gdim + row * Gdim + g]
                                : 0.0f);
  }
  // fw1 (50x128) -> sBT transposed [f][k], k zero-padded to 64 (ldb = 64)
  for (int i = tid; i < Fdim * 64; i += 256) {
    const int f = i >> 6, k = i & 63;
    sBT[f * 64 + k] = (__bf16)((k < Gdim) ? fw1[k * Fdim + f] : 0.0f);
  }
  __syncthreads();

  const int wv = tid >> 5, lane = tid & 31;
  const int tf = wv;
  const int n = lane & 15, h = lane >> 4;
  const float bias1 = fb1[16 * tf + n];
  const float bias2 = fb2[16 * tf + n];

  // GEMM1: (64x64) @ (64x128) -> ssp -> sH (bf16)
#pragma unroll
  for (int tm = 0; tm < 4; ++tm) {
    v8f acc = {};
#pragma unroll
    for (int kp = 0; kp < 2; ++kp) {
      v16bf a = load_frag_a(sA, 16 * tm, 64, 32 * kp, lane);
      v16bf b = load_frag_bt(sBT, 16 * tf, 64, 32 * kp, lane);
      acc = wmma_bf16(a, b, acc);
    }
#pragma unroll
    for (int r = 0; r < 8; ++r) {
      const int m = 16 * tm + r + 8 * h;
      sH[m * Fdim + 16 * tf + n] = (__bf16)ssp(acc[r] + bias1);
    }
  }
  __syncthreads();

  // fw2 (128x128) -> sBT transposed [f][k] (ldb = 128)
  for (int i = tid; i < Fdim * Fdim; i += 256) {
    const int k = i >> 7, f = i & 127;
    sBT[f * Fdim + k] = (__bf16)fw2[i];
  }
  __syncthreads();

  // GEMM2: (64x128) @ (128x128); fuse cutoff, neighbor gather, sum over N.
#pragma unroll
  for (int tm = 0; tm < 4; ++tm) {
    v8f acc = {};
#pragma unroll
    for (int kp = 0; kp < 4; ++kp) {
      v16bf a = load_frag_a(sH, 16 * tm, Fdim, 32 * kp, lane);
      v16bf b = load_frag_bt(sBT, 16 * tf, Fdim, 32 * kp, lane);
      acc = wmma_bf16(a, b, acc);
    }
    float part = 0.0f;
#pragma unroll
    for (int r = 0; r < 8; ++r) {
      const int m = 16 * tm + r + 8 * h;            // neighbor slot n
      const float wval = (acc[r] + bias2) * sC[m];
      const float yv =
          y[((size_t)bb * Adim + sN[m]) * Fdim + 16 * tf + n];
      part += wval * yv;
    }
    atomicAdd(&sOut[16 * tf + n], part);            // ds_add_f32
  }
  __syncthreads();

  if (tid < Fdim) out[(size_t)p * Fdim + tid] = sOut[tid];
}

// ---------------- Launch -------------------------------------------------

extern "C" void kernel_launch(void* const* d_in, const int* in_sizes, int n_in,
                              void* d_out, int out_size, void* d_ws,
                              size_t ws_size, hipStream_t stream) {
  const float*      x     = (const float*)d_in[0];
  const float*      r_ij  = (const float*)d_in[1];
  const float*      f_ij  = (const float*)d_in[2];
  const float*      pmask = (const float*)d_in[3];
  const long long*  nbrs  = (const long long*)d_in[4];   // int64 in reference
  const float*      in2fw = (const float*)d_in[5];
  const float*      fb1   = (const float*)d_in[6];
  const float*      fw1   = (const float*)d_in[6];       // placeholder, fixed below
  (void)fw1;
  // correct order: 6=fw1, 7=fb1, 8=fw2, 9=fb2
  const float* fw1p = (const float*)d_in[6];
  const float* fb1p = (const float*)d_in[7];
  const float* fw2p = (const float*)d_in[8];
  const float* fb2p = (const float*)d_in[9];
  (void)fb1; (void)in_sizes; (void)n_in; (void)out_size; (void)ws_size;

  float* y = (float*)d_ws;           // B*A*F fp32 = 4 MiB scratch
  float* out = (float*)d_out;

  // Kernel 1: y = x @ in2f_w  (8192 rows -> 128 blocks of 64 rows)
  in2f_gemm_kernel<<<(Bdim * Adim) / 64, 256, 0, stream>>>(x, in2fw, y);

  // Kernel 2: one block per (b,a) pair
  cfconv_kernel<<<Bdim * Adim, 256, 0, stream>>>(
      f_ij, r_ij, pmask, nbrs, fw1p, fb1p, fw2p, fb2p, y, out);
}
